// PointNetSetAbstraction_23149873725517
// MI455X (gfx1250) — matrix operand order repaired and checked by reading, not compiled, not run on hardware
//
#include <hip/hip_runtime.h>
#include <hip/hip_bf16.h>

#define NB       8
#define NPTS     16384
#define NPOINT   2048
#define NSAMPLE  32
#define RAD2     0.01f
#define CIN      16
#define H3       64

typedef __attribute__((ext_vector_type(16))) _Float16 v16h;
typedef __attribute__((ext_vector_type(8)))  _Float16 v8h;
typedef __attribute__((ext_vector_type(8)))  float    v8f;

// ---------------------------------------------------------------------------
// Kernel 1: farthest point sampling. One block per batch; the whole point
// cloud + running min-distance live in registers (16 pts/thread).
// Writes new_xyz (B, NPOINT, 3) directly into d_out.
// ---------------------------------------------------------------------------
__global__ __launch_bounds__(1024) void fps_kernel(const float* __restrict__ xyz,
                                                   float* __restrict__ new_xyz) {
    const int b    = blockIdx.x;
    const int tid  = threadIdx.x;
    const int lane = tid & 31;
    const int wid  = tid >> 5;
    const float* base = xyz + (size_t)b * NPTS * 3;

    float px[16], py[16], pz[16], mind[16];
#pragma unroll
    for (int j = 0; j < 16; ++j) {
        int p = j * 1024 + tid;
        px[j] = base[p * 3 + 0];
        py[j] = base[p * 3 + 1];
        pz[j] = base[p * 3 + 2];
        mind[j] = 1e10f;
    }

    __shared__ float rv[32];
    __shared__ int   rp[32];
    __shared__ float cc[3];
    __shared__ int   curs;

    int cur = 0;
    for (int it = 0; it < NPOINT; ++it) {
        // owner thread of point `cur` publishes the centroid coords
        if ((cur & 1023) == tid) {
            int j = cur >> 10;
#pragma unroll
            for (int jj = 0; jj < 16; ++jj)
                if (jj == j) { cc[0] = px[jj]; cc[1] = py[jj]; cc[2] = pz[jj]; }
        }
        __syncthreads();
        float bx = cc[0], by = cc[1], bz = cc[2];
        if (tid == 0) {
            float* o = new_xyz + ((size_t)b * NPOINT + it) * 3;
            o[0] = bx; o[1] = by; o[2] = bz;
        }

        float bestv = -1.0f;
        int   bestp = 0;
#pragma unroll
        for (int j = 0; j < 16; ++j) {
            float dx = px[j] - bx, dy = py[j] - by, dz = pz[j] - bz;
            float d  = dx * dx + dy * dy + dz * dz;
            float md = fminf(mind[j], d);
            mind[j] = md;
            int p = j * 1024 + tid;
            if (md > bestv || (md == bestv && p < bestp)) { bestv = md; bestp = p; }
        }
        // wave32 argmax reduce (first-max tie break = lowest index, matches jnp.argmax)
#pragma unroll
        for (int off = 16; off >= 1; off >>= 1) {
            float ov = __shfl_xor(bestv, off);
            int   op = __shfl_xor(bestp, off);
            if (ov > bestv || (ov == bestv && op < bestp)) { bestv = ov; bestp = op; }
        }
        if (lane == 0) { rv[wid] = bestv; rp[wid] = bestp; }
        __syncthreads();
        if (wid == 0) {
            float v = rv[lane];
            int   p = rp[lane];
#pragma unroll
            for (int off = 16; off >= 1; off >>= 1) {
                float ov = __shfl_xor(v, off);
                int   op = __shfl_xor(p, off);
                if (ov > v || (ov == v && op < p)) { v = ov; p = op; }
            }
            if (lane == 0) curs = p;
        }
        __syncthreads();
        cur = curs;
    }
}

// ---------------------------------------------------------------------------
// Kernel 2: ball query. One wave per centroid (nsample == warpSize == 32).
// Picks the first 32 in-radius indices in ascending order (exact early exit),
// pads with the first hit — identical to the reference top_k construction.
// ---------------------------------------------------------------------------
__global__ __launch_bounds__(256) void ballq_kernel(const float* __restrict__ xyz,
                                                    const float* __restrict__ new_xyz,
                                                    int* __restrict__ idx) {
    const int wgid = blockIdx.x * 8 + (threadIdx.x >> 5);
    const int lane = threadIdx.x & 31;
    const int b = wgid / NPOINT;
    const int s = wgid % NPOINT;

    const float* q = new_xyz + ((size_t)b * NPOINT + s) * 3;
    const float qx = q[0], qy = q[1], qz = q[2];
    const float* base = xyz + (size_t)b * NPTS * 3;

    int  cnt = 0;
    int  my  = 0;
    bool have = false;
    for (int p0 = 0; p0 < NPTS && cnt < NSAMPLE; p0 += 32) {
        int p = p0 + lane;
        float dx = base[p * 3 + 0] - qx;
        float dy = base[p * 3 + 1] - qy;
        float dz = base[p * 3 + 2] - qz;
        float d2 = dx * dx + dy * dy + dz * dz;
        unsigned m = (unsigned)__ballot(d2 <= RAD2);
        int c = __popc(m);
        int k = lane - cnt;                 // which hit of this chunk is mine
        if (!have && k >= 0 && k < c) {
            unsigned mm = m;
#pragma unroll
            for (int j = 0; j < 31; ++j)
                if (j < k) mm &= mm - 1;    // drop k lowest set bits
            my = p0 + __ffs(mm) - 1;
            have = true;
        }
        cnt += c;
    }
    int first = __shfl(my, 0);              // centroid itself always qualifies
    if (!have) my = first;
    idx[(size_t)wgid * NSAMPLE + lane] = my;
}

// ---------------------------------------------------------------------------
// Kernel 3: grouped MLP via v_wmma_f32_16x16x32_f16 + max pooling.
// One wave per centroid (grouped block = 32x16 A-matrix), weights cached in
// registers as WMMA B-tiles, relayout between layers via LDS staging.
// ---------------------------------------------------------------------------
#define WAVES3 8
#define CPW    4

__device__ inline v16h load_b_tile(const float* __restrict__ W, int Nw, int Kvalid,
                                   int n0, int lane) {
    // B 32x16 in this code's contiguous k order: VGPR v -> rows K = 2v + 16*(lane>=16) (+1 in hi half)
    v16h r;
    int col = n0 + (lane & 15);
    int kb  = (lane & 16) ? 16 : 0;
#pragma unroll
    for (int v = 0; v < 8; ++v) {
        int k0 = kb + 2 * v;
        float f0 = (k0     < Kvalid) ? W[(size_t)k0 * Nw + col]       : 0.0f;
        float f1 = (k0 + 1 < Kvalid) ? W[(size_t)(k0 + 1) * Nw + col] : 0.0f;
        r[2 * v]     = (_Float16)f0;
        r[2 * v + 1] = (_Float16)f1;
    }
    return r;
}

__device__ inline v16h load_a_tile(const _Float16* stage, int mi, int lane) {
    // A 16x32 in contiguous k order (same as B here): lanes 0-15 row M=l hold K0-15; lanes 16-31 row M=l-16
    // hold K16-31  ->  two 16B-aligned ds_load_b128 from row-major [32][32].
    int row = mi * 16 + (lane & 15);
    int kA  = (lane & 16) ? 16 : 0;
    v8h lo = *(const v8h*)(stage + row * 32 + kA);
    v8h hi = *(const v8h*)(stage + row * 32 + kA + 8);
    v16h a;
#pragma unroll
    for (int i = 0; i < 8; ++i) { a[i] = lo[i]; a[i + 8] = hi[i]; }
    return a;
}

__device__ inline void store_h_tile(_Float16* stage, int mi, int ni, v8f acc,
                                    int lane, bool relu) {
    // C layout: VGPR r, lane l -> (M = mi*16 + r + 8*(l>=16), N = ni*16 + l%16)
    int col = ni * 16 + (lane & 15);
    int rb  = mi * 16 + ((lane & 16) ? 8 : 0);
#pragma unroll
    for (int r = 0; r < 8; ++r) {
        float x = acc[r];
        if (relu) x = fmaxf(x, 0.0f);
        stage[(rb + r) * 32 + col] = (_Float16)x;
    }
}

__device__ inline v8f splat8(float x) {
    v8f c;
#pragma unroll
    for (int i = 0; i < 8; ++i) c[i] = x;
    return c;
}

__global__ __launch_bounds__(256) void mlp_kernel(const float* __restrict__ features,
                                                  const int* __restrict__ idx,
                                                  const float* __restrict__ W1,
                                                  const float* __restrict__ b1,
                                                  const float* __restrict__ W2,
                                                  const float* __restrict__ b2,
                                                  const float* __restrict__ W3,
                                                  const float* __restrict__ b3,
                                                  float* __restrict__ out_feat) {
    __shared__ __align__(16) _Float16 stage_all[WAVES3][32][32];
    const int w    = threadIdx.x >> 5;
    const int lane = threadIdx.x & 31;
    _Float16* stage = &stage_all[w][0][0];
    const int wg = blockIdx.x * WAVES3 + w;

    // Weights as B-tiles in registers (layer-1 K padded 16 -> 32 with zeros).
    v16h B1[2], B2[2], B3[4];
#pragma unroll
    for (int ni = 0; ni < 2; ++ni) B1[ni] = load_b_tile(W1, 32, 16, ni * 16, lane);
#pragma unroll
    for (int ni = 0; ni < 2; ++ni) B2[ni] = load_b_tile(W2, 32, 32, ni * 16, lane);
#pragma unroll
    for (int ni = 0; ni < 4; ++ni) B3[ni] = load_b_tile(W3, 64, 32, ni * 16, lane);

    float bias1[2], bias2[2], bias3[4];
#pragma unroll
    for (int ni = 0; ni < 2; ++ni) bias1[ni] = b1[ni * 16 + (lane & 15)];
#pragma unroll
    for (int ni = 0; ni < 2; ++ni) bias2[ni] = b2[ni * 16 + (lane & 15)];
#pragma unroll
    for (int ni = 0; ni < 4; ++ni) bias3[ni] = b3[ni * 16 + (lane & 15)];

    for (int ci = 0; ci < CPW; ++ci) {
        const int cent = wg * CPW + ci;
        const int b = cent / NPOINT;
        const int s = cent % NPOINT;

        // Gather: lane k owns neighbor k -> row k of A staging (K 16..23 zeroed; W1 tile is zero for K>=16).
        int nk = idx[(size_t)cent * NSAMPLE + lane];
        const float4* fp = (const float4*)(features + ((size_t)b * NPTS + nk) * CIN);
        float4 f0 = fp[0], f1 = fp[1], f2 = fp[2], f3 = fp[3];
        v8h g0, g1;
        g0[0] = (_Float16)f0.x; g0[1] = (_Float16)f0.y; g0[2] = (_Float16)f0.z; g0[3] = (_Float16)f0.w;
        g0[4] = (_Float16)f1.x; g0[5] = (_Float16)f1.y; g0[6] = (_Float16)f1.z; g0[7] = (_Float16)f1.w;
        g1[0] = (_Float16)f2.x; g1[1] = (_Float16)f2.y; g1[2] = (_Float16)f2.z; g1[3] = (_Float16)f2.w;
        g1[4] = (_Float16)f3.x; g1[5] = (_Float16)f3.y; g1[6] = (_Float16)f3.z; g1[7] = (_Float16)f3.w;
        _Float16* row = stage + lane * 32;
        *(v8h*)(row)      = g0;
        *(v8h*)(row + 8)  = g1;
        *(uint4*)(row + 16) = uint4{0, 0, 0, 0};
        __syncthreads();

        // Layer 1: (32x32pad) @ (32x32) + b1, relu
        v16h a0 = load_a_tile(stage, 0, lane);
        v16h a1 = load_a_tile(stage, 1, lane);
        v8f c1[2][2];
#pragma unroll
        for (int ni = 0; ni < 2; ++ni) {
            c1[0][ni] = splat8(bias1[ni]);
            c1[1][ni] = splat8(bias1[ni]);
            c1[0][ni] = __builtin_amdgcn_wmma_f32_16x16x32_f16(false, a0, false, B1[ni], (short)0, c1[0][ni], false, false);
            c1[1][ni] = __builtin_amdgcn_wmma_f32_16x16x32_f16(false, a1, false, B1[ni], (short)0, c1[1][ni], false, false);
        }
        __syncthreads();
#pragma unroll
        for (int mi = 0; mi < 2; ++mi)
#pragma unroll
            for (int ni = 0; ni < 2; ++ni)
                store_h_tile(stage, mi, ni, c1[mi][ni], lane, true);
        __syncthreads();

        // Layer 2: (32x32) @ (32x32) + b2, relu
        a0 = load_a_tile(stage, 0, lane);
        a1 = load_a_tile(stage, 1, lane);
        v8f c2[2][2];
#pragma unroll
        for (int ni = 0; ni < 2; ++ni) {
            c2[0][ni] = splat8(bias2[ni]);
            c2[1][ni] = splat8(bias2[ni]);
            c2[0][ni] = __builtin_amdgcn_wmma_f32_16x16x32_f16(false, a0, false, B2[ni], (short)0, c2[0][ni], false, false);
            c2[1][ni] = __builtin_amdgcn_wmma_f32_16x16x32_f16(false, a1, false, B2[ni], (short)0, c2[1][ni], false, false);
        }
        __syncthreads();
#pragma unroll
        for (int mi = 0; mi < 2; ++mi)
#pragma unroll
            for (int ni = 0; ni < 2; ++ni)
                store_h_tile(stage, mi, ni, c2[mi][ni], lane, true);
        __syncthreads();

        // Layer 3: (32x32) @ (32x64) + b3 (no relu)
        a0 = load_a_tile(stage, 0, lane);
        a1 = load_a_tile(stage, 1, lane);
        v8f c3[2][4];
#pragma unroll
        for (int ni = 0; ni < 4; ++ni) {
            c3[0][ni] = splat8(bias3[ni]);
            c3[1][ni] = splat8(bias3[ni]);
            c3[0][ni] = __builtin_amdgcn_wmma_f32_16x16x32_f16(false, a0, false, B3[ni], (short)0, c3[0][ni], false, false);
            c3[1][ni] = __builtin_amdgcn_wmma_f32_16x16x32_f16(false, a1, false, B3[ni], (short)0, c3[1][ni], false, false);
        }

        // Max over the 32 neighbors (rows M): in-lane over 8 regs x 2 M-tiles,
        // then fold lanes l <-> l+16 (upper/lower row halves of same column).
#pragma unroll
        for (int ni = 0; ni < 4; ++ni) {
            float m = -3.402823466e+38f;
#pragma unroll
            for (int mi = 0; mi < 2; ++mi)
#pragma unroll
                for (int r = 0; r < 8; ++r) m = fmaxf(m, c3[mi][ni][r]);
            m = fmaxf(m, __shfl_xor(m, 16));
            if (lane < 16)
                out_feat[((size_t)b * NPOINT + s) * H3 + ni * 16 + lane] = m;
        }
        __syncthreads();
    }
}

// ---------------------------------------------------------------------------
extern "C" void kernel_launch(void* const* d_in, const int* in_sizes, int n_in,
                              void* d_out, int out_size, void* d_ws, size_t ws_size,
                              hipStream_t stream) {
    const float* xyz      = (const float*)d_in[0];
    const float* features = (const float*)d_in[1];
    const float* W1 = (const float*)d_in[2];
    const float* b1 = (const float*)d_in[3];
    const float* W2 = (const float*)d_in[4];
    const float* b2 = (const float*)d_in[5];
    const float* W3 = (const float*)d_in[6];
    const float* b3 = (const float*)d_in[7];

    float* new_xyz  = (float*)d_out;                         // (B, NPOINT, 3)
    float* new_feat = new_xyz + (size_t)NB * NPOINT * 3;     // (B, NPOINT, 64)
    int*   idx      = (int*)d_ws;                            // (B*NPOINT, 32) = 2 MB

    fps_kernel  <<<NB, 1024, 0, stream>>>(xyz, new_xyz);
    ballq_kernel<<<NB * NPOINT / 8, 256, 0, stream>>>(xyz, new_xyz, idx);
    mlp_kernel  <<<NB * NPOINT / (WAVES3 * CPW), 256, 0, stream>>>(
        features, idx, W1, b1, W2, b2, W3, b3, new_feat);
}
